// SingleHeadAttention_36644660970252
// MI455X (gfx1250) — matrix operand order repaired and checked by reading, not compiled, not run on hardware
//
#include <hip/hip_runtime.h>
#include <hip/hip_bf16.h>

typedef __attribute__((ext_vector_type(16))) _Float16 v16h;
typedef __attribute__((ext_vector_type(8)))  _Float16 v8h;
typedef __attribute__((ext_vector_type(8)))  float    v8f;
typedef __attribute__((ext_vector_type(4)))  float    v4f;

#define BATCH 32
#define SEQ   2048
#define DH    8
#define QT    16      // queries per wave
#define KT    32      // keys per iteration
// (1/sqrt(8)) * log2(e): logits produced directly in exp2 domain
#define QSCALE (0.35355339059327373f * 1.4426950408889634f)

// ---------------------------------------------------------------------------
// Pre-pass: Q/K/V linear projections (torch Linear: x @ W^T + b), f32 -> f16.
// Q is pre-scaled by (1/sqrt(D))*log2(e) so softmax uses raw v_exp_f32 (exp2).
// V is written TRANSPOSED as Vt[b][d][s] for contiguous V^T A-tile loads.
// ---------------------------------------------------------------------------
__global__ __launch_bounds__(256) void proj_qkv_kernel(
    const float* __restrict__ x,
    const float* __restrict__ Wq, const float* __restrict__ bq,
    const float* __restrict__ Wk, const float* __restrict__ bk,
    const float* __restrict__ Wv, const float* __restrict__ bv,
    _Float16* __restrict__ Qh,    // [B][S][8]  (scaled)
    _Float16* __restrict__ Kh,    // [B][S][8]
    _Float16* __restrict__ Vt)    // [B][8][S]
{
  const int idx = blockIdx.x * blockDim.x + threadIdx.x;   // row over B*S
  if (idx >= BATCH * SEQ) return;
  const int b = idx >> 11;
  const int s = idx & (SEQ - 1);

  const v4f x0 = *(const v4f*)(x + (size_t)idx * DH);
  const v4f x1 = *(const v4f*)(x + (size_t)idx * DH + 4);
  float xi[DH] = {x0[0], x0[1], x0[2], x0[3], x1[0], x1[1], x1[2], x1[3]};

  v8h qrow, krow;
  float vvals[DH];
#pragma unroll
  for (int e = 0; e < DH; ++e) {
    float aq = bq[e], ak = bk[e], av = bv[e];
#pragma unroll
    for (int d = 0; d < DH; ++d) {
      aq = fmaf(xi[d], Wq[e * DH + d], aq);
      ak = fmaf(xi[d], Wk[e * DH + d], ak);
      av = fmaf(xi[d], Wv[e * DH + d], av);
    }
    qrow[e] = (_Float16)(aq * QSCALE);
    krow[e] = (_Float16)ak;
    vvals[e] = av;
  }
  *(v8h*)(Qh + (size_t)idx * DH) = qrow;
  *(v8h*)(Kh + (size_t)idx * DH) = krow;
#pragma unroll
  for (int e = 0; e < DH; ++e)
    Vt[((size_t)(b * DH + e)) * SEQ + s] = (_Float16)vvals[e];
}

// ---------------------------------------------------------------------------
// Flash attention, one 16-query tile per wave32.
//   S^T = K · Q^T     (v_wmma_f32_16x16x32_f16, K-dim = 8 padded to 32)
//   O^T = V^T · P^T   (v_wmma_f32_16x16x32_f16, K-dim = 32 keys, full)
// V^T carries a ones-row at d=8 so O^T row 8 accumulates the softmax
// denominator under the identical corr-rescale recurrence (no VALU row-sum).
// ---------------------------------------------------------------------------
__global__ __launch_bounds__(256) void flash_attn_kernel(
    const _Float16* __restrict__ Qh,   // [B][S][8]  pre-scaled (exp2 domain)
    const _Float16* __restrict__ Kh,   // [B][S][8]
    const _Float16* __restrict__ Vt,   // [B][8][S]
    const float* __restrict__ Wo,      // [8][8]
    const float* __restrict__ bo,      // [8]
    float* __restrict__ out)           // [B][S][8]
{
  const int lane = threadIdx.x & 31;
  const int wave = threadIdx.x >> 5;
  const int tile = blockIdx.x * 8 + wave;      // 4096 tiles total
  const int b    = tile >> 7;                  // / (SEQ/QT)
  const int qt   = tile & 127;
  const int qbase = qt * QT;

  const bool lo  = lane < 16;
  const int  l16 = lane & 15;

  // --- Q as B-matrix: lane<16 holds column q=l16, d=0..7 in elements 0..7 ---
  v16h qb = {};
  if (lo) {
    const v8h qr = *(const v8h*)(Qh + ((size_t)b * SEQ + qbase + l16) * DH);
#pragma unroll
    for (int j = 0; j < 8; ++j) qb[j] = qr[j];
  }

  v8f   o = {};           // O^T acc: lane<16 vgpr r -> O[q=l16][d=r];
                          // lane>=16 vgpr 0 -> softmax denominator row (d=8)
  float m = -1e30f;       // running max (exp2 domain) for this lane's query
  const v8f zc = {};

  const _Float16* kbase = Kh + (size_t)b * SEQ * DH;
  const _Float16* vbase = Vt + (size_t)b * DH * SEQ;

#pragma unroll 2
  for (int kt = 0; kt < SEQ; kt += KT) {
    // --- K A-tiles (rows = keys, K-dim = d, only lanes<16 carry data) ---
    v16h ka0 = {}, ka1 = {};
    if (lo) {
      const v8h r0 = *(const v8h*)(kbase + (size_t)(kt + l16) * DH);
      const v8h r1 = *(const v8h*)(kbase + (size_t)(kt + 16 + l16) * DH);
#pragma unroll
      for (int j = 0; j < 8; ++j) { ka0[j] = r0[j]; ka1[j] = r1[j]; }
    }

    // --- V^T A-tile (rows = d, K-dim = 32 keys) + ones row at d=8 ---
    v16h va = {};
    if (l16 < DH) {
      const _Float16* vrow = vbase + (size_t)l16 * SEQ + kt;
      const v8h v0 = *(const v8h*)(vrow + (lo ? 0 : 16));
      const v8h v1 = *(const v8h*)(vrow + (lo ? 8 : 24));
#pragma unroll
      for (int j = 0; j < 8; ++j) { va[j] = v0[j]; va[8 + j] = v1[j]; }
    } else if (l16 == DH) {
#pragma unroll
      for (int j = 0; j < 16; ++j) va[j] = (_Float16)1.0f;
    }

    // --- S^T tiles: keys kt..kt+15 and kt+16..kt+31 (exp2-domain logits) ---
    v8f st0 = __builtin_amdgcn_wmma_f32_16x16x32_f16(false, ka0, false, qb,
                                                     (short)0, zc, false, false);
    v8f st1 = __builtin_amdgcn_wmma_f32_16x16x32_f16(false, ka1, false, qb,
                                                     (short)0, zc, false, false);

    // --- online softmax max update (per-query stats live in each lane) ---
    float tm = -1e30f;
#pragma unroll
    for (int r = 0; r < 8; ++r) tm = fmaxf(tm, fmaxf(st0[r], st1[r]));
    tm = fmaxf(tm, __shfl_xor(tm, 16, 32));
    const float mn   = fmaxf(m, tm);
    const float corr = __builtin_amdgcn_exp2f(m - mn);
    m = mn;
#pragma unroll
    for (int r = 0; r < 8; ++r) o[r] *= corr;   // also rescales denom row

    // --- p = exp2(s - m), packed straight to f16 pairs ---
    union H4 { v8h v; unsigned int u[4]; } hp0, hp1;
#pragma unroll
    for (int r = 0; r < 8; ++r) {
      hp0.v[r] = (_Float16)__builtin_amdgcn_exp2f(st0[r] - mn);
      hp1.v[r] = (_Float16)__builtin_amdgcn_exp2f(st1[r] - mn);
    }

    // --- build P^T B-matrix with 4 xor-16 shuffles of packed pairs ---
    // lane<16 needs partner's p0 (keys 8..15); lane>=16 needs partner's p1
    // (keys 16..23) -> each lane sends (lo ? p1 : p0), one shfl per word.
    union PB { v16h v; unsigned int u[8]; } pb;
#pragma unroll
    for (int j = 0; j < 4; ++j) {
      const unsigned int snd = lo ? hp1.u[j] : hp0.u[j];
      const unsigned int rcv = __shfl_xor(snd, 16, 32);
      pb.u[j]     = lo ? hp0.u[j] : rcv;   // keys 0..7  | 16..23
      pb.u[4 + j] = lo ? rcv : hp1.u[j];   // keys 8..15 | 24..31
    }

    // --- O^T += V^T · P^T  (row 8 accumulates the denominator) ---
    o = __builtin_amdgcn_wmma_f32_16x16x32_f16(false, va, false, pb.v,
                                               (short)0, o, false, false);
  }

  // --- epilogue: fetch denominator, normalize, output projection, store ---
  const float ls = __shfl_xor(o[0], 16, 32);   // lane<16 <- partner row d=8
  if (lo) {
    const float inv = 1.0f / ls;
    float od[DH];
#pragma unroll
    for (int d = 0; d < DH; ++d) od[d] = o[d] * inv;

    float res[DH];
#pragma unroll
    for (int e = 0; e < DH; ++e) {
      float acc = bo[e];
#pragma unroll
      for (int d = 0; d < DH; ++d) acc = fmaf(od[d], Wo[e * DH + d], acc);
      res[e] = acc;
    }
    float* op = out + ((size_t)b * SEQ + qbase + l16) * DH;
    v4f r0 = {res[0], res[1], res[2], res[3]};
    v4f r1 = {res[4], res[5], res[6], res[7]};
    *(v4f*)op       = r0;
    *((v4f*)op + 1) = r1;
  }
}

// ---------------------------------------------------------------------------
extern "C" void kernel_launch(void* const* d_in, const int* in_sizes, int n_in,
                              void* d_out, int out_size, void* d_ws, size_t ws_size,
                              hipStream_t stream) {
  const float* x  = (const float*)d_in[0];
  const float* Wq = (const float*)d_in[1];
  const float* bq = (const float*)d_in[2];
  const float* Wk = (const float*)d_in[3];
  const float* bk = (const float*)d_in[4];
  const float* Wv = (const float*)d_in[5];
  const float* bv = (const float*)d_in[6];
  const float* Wo = (const float*)d_in[7];
  const float* bo = (const float*)d_in[8];

  _Float16* Qh = (_Float16*)d_ws;                       // 1 MB
  _Float16* Kh = Qh + (size_t)BATCH * SEQ * DH;         // 1 MB
  _Float16* Vt = Kh + (size_t)BATCH * SEQ * DH;         // 1 MB

  const int rows = BATCH * SEQ;                         // 65536
  proj_qkv_kernel<<<rows / 256, 256, 0, stream>>>(x, Wq, bq, Wk, bk, Wv, bv,
                                                  Qh, Kh, Vt);

  // 4096 query tiles, 8 waves (tiles) per 256-thread block -> 512 blocks
  flash_attn_kernel<<<512, 256, 0, stream>>>(Qh, Kh, Vt, Wo, bo, (float*)d_out);
}
